// Wav2Vec2GumbelVectorQuantizer_50938312131005
// MI455X (gfx1250) — matrix operand order repaired and checked by reading, not compiled, not run on hardware
//
#include <hip/hip_runtime.h>
#include <stdint.h>

// ---------------------------------------------------------------------------
// Wav2Vec2 Gumbel Vector Quantizer, MI455X (gfx1250, wave32, WMMA + async LDS).
//
// N=32768 rows, H=512, G=2, V=320 (GV=640), Dg=128.
// Round-2 changes vs round 1:
//  * W pre-converted once to bf16, transposed [col][k] in workspace (prep
//    kernel) -- kills the per-K-step f32->bf16 staging flood.
//  * Per-K-step W slice staged into LDS with GLOBAL_LOAD_ASYNC_TO_LDS_B128
//    (ASYNCcnt), double-buffered so DMA of slice k+1 overlaps WMMAs of k.
//  * B fragments double-buffered in VGPRs -> WMMAs wait dscnt<=2 instead of 0.
//  * A-fragment raw data prefetched one K-step ahead.
// Workspace: margPart 1024*640 f32 (2.62MB) + Wt bf16 640*512 (0.66MB).
// ---------------------------------------------------------------------------

typedef __attribute__((ext_vector_type(16))) __bf16 v16bf;
typedef __attribute__((ext_vector_type(8)))  float  v8f;

#define N_ROWS  32768
#define H_DIM   512
#define GV      640
#define V_CODES 320
#define DG      128
#define NBLK    (N_ROWS / 32)   // 1024 main-kernel blocks
#define BPITCH  40              // LDS pitch (ushorts) per column: 80B
#define EPSF    1e-7f

union BF16x16 {
    v16bf v;
    unsigned short u[16];
    uint4 q[2];
};

static __device__ inline unsigned short f32_to_bf16_bits(float f) {
    __bf16 h = (__bf16)f;                       // RNE convert (v_cvt)
    return __builtin_bit_cast(unsigned short, h);
}

// ---------------------------------------------------------------------------
// Prep: W [512,640] f32 -> Wt [640,512] bf16 (transposed, contiguous in K).
// ---------------------------------------------------------------------------
__global__ __launch_bounds__(256)
void prep_w(const float* __restrict__ W, unsigned short* __restrict__ Wt) {
    int i = blockIdx.x * 256 + threadIdx.x;     // 327680 elems
    int k = i / GV;
    int c = i - k * GV;
    Wt[(size_t)c * H_DIM + k] = f32_to_bf16_bits(W[i]);
}

// ---------------------------------------------------------------------------
// Main fused kernel. Block: 128 threads = 4 waves = {row tile} x {group}.
// Each wave: 16 rows x 320 cols (20 WMMA col tiles), K=512 in 16 steps.
// ---------------------------------------------------------------------------
__global__ __launch_bounds__(128)
void gumbel_vq_main(const float* __restrict__ hs,      // [N, 512]
                    const int*   __restrict__ mask,    // [N]
                    const float* __restrict__ gum,     // [N, 640]
                    const unsigned short* __restrict__ Wt, // [640, 512] bf16
                    const float* __restrict__ bias,    // [640]
                    const float* __restrict__ cvs,     // [2, 320, 128]
                    float* __restrict__ out,           // [N, 256] (+ppl slot)
                    float* __restrict__ margPart)      // [NBLK, 640]
{
    const int tid  = threadIdx.x;
    const int wave = tid >> 5;
    const int lane = tid & 31;
    const int half = lane >> 4;
    const int l16  = lane & 15;
    const int rsel = wave & 1;        // row tile within block
    const int g    = wave >> 1;       // codebook group
    const int rowBase = blockIdx.x * 32 + rsel * 16;
    const int colBase = g * V_CODES;

    // double-buffered W slice: [buf][col][k_local], bf16, pitch 40 ushorts
    __shared__ __attribute__((aligned(16))) unsigned short Bsh[2][GV * BPITCH];
    __shared__ float margLDS[GV];

    v8f c[20] = {};

    // ---- async-stage helper (20 x b128 per thread): cols tid+128j, 64B each
    #define STAGE_SLICE(bufsel, kkv)                                           \
        do {                                                                   \
            _Pragma("unroll")                                                  \
            for (int j = 0; j < 5; ++j) {                                      \
                int cc = tid + j * 128;                                        \
                unsigned voff = (unsigned)((cc * H_DIM + (kkv)) * 2);          \
                unsigned ldsa = (unsigned)(uintptr_t)(&Bsh[bufsel][cc * BPITCH]); \
                asm volatile(                                                  \
                    "global_load_async_to_lds_b128 %0, %1, %2 offset:0\n\t"    \
                    "global_load_async_to_lds_b128 %0, %1, %2 offset:16\n\t"   \
                    "global_load_async_to_lds_b128 %0, %1, %2 offset:32\n\t"   \
                    "global_load_async_to_lds_b128 %0, %1, %2 offset:48"       \
                    :: "v"(ldsa), "v"(voff), "s"(Wt) : "memory");              \
            }                                                                  \
        } while (0)

    // ---- prologue: stage slice 0, prefetch A raw for kk=0
    STAGE_SLICE(0, 0);

    const float* arow = hs + (size_t)(rowBase + l16) * H_DIM;
    float4 ar0 = *(const float4*)(arow + half * 8);
    float4 ar1 = *(const float4*)(arow + half * 8 + 4);
    float4 ar2 = *(const float4*)(arow + 16 + half * 8);
    float4 ar3 = *(const float4*)(arow + 16 + half * 8 + 4);

    asm volatile("s_wait_asynccnt 0x0" ::: "memory");
    __syncthreads();

    for (int kk16 = 0; kk16 < 16; ++kk16) {
        const int kk  = kk16 * 32;
        const int cur = kk16 & 1;
        const int nxt = cur ^ 1;

        // ---- kick async DMA of next slice (overlaps with WMMAs below)
        if (kk16 < 15) STAGE_SLICE(nxt, kk + 32);

        // ---- pack A fragment (16x32 bf16) from prefetched raw data
        BF16x16 a;
        {
            const float fl[16] = { ar0.x,ar0.y,ar0.z,ar0.w, ar1.x,ar1.y,ar1.z,ar1.w,
                                   ar2.x,ar2.y,ar2.z,ar2.w, ar3.x,ar3.y,ar3.z,ar3.w };
            #pragma unroll
            for (int i = 0; i < 16; ++i) a.u[i] = f32_to_bf16_bits(fl[i]);
        }
        // prefetch A raw for next K-step
        if (kk16 < 15) {
            ar0 = *(const float4*)(arow + kk + 32 + half * 8);
            ar1 = *(const float4*)(arow + kk + 32 + half * 8 + 4);
            ar2 = *(const float4*)(arow + kk + 48 + half * 8);
            ar3 = *(const float4*)(arow + kk + 48 + half * 8 + 4);
        }

        // ---- 20 WMMAs, B fragments double-buffered in VGPRs
        BF16x16 b[2];
        {
            const uint4* bp = (const uint4*)
                (&Bsh[cur][(colBase + l16) * BPITCH + half * 8]);
            b[0].q[0] = bp[0];
            b[0].q[1] = bp[2];
        }
        #pragma unroll
        for (int t = 0; t < 20; ++t) {
            if (t < 19) {
                const uint4* bp = (const uint4*)
                    (&Bsh[cur][(colBase + (t + 1) * 16 + l16) * BPITCH + half * 8]);
                b[(t + 1) & 1].q[0] = bp[0];
                b[(t + 1) & 1].q[1] = bp[2];
            }
            c[t] = __builtin_amdgcn_wmma_f32_16x16x32_bf16(
                       false, a.v, false, b[t & 1].v, (short)0, c[t], false, false);
        }

        // ---- close the double-buffer handoff
        if (kk16 < 15) {
            asm volatile("s_wait_asynccnt 0x0" ::: "memory");
            __syncthreads();
        }
    }
    #undef STAGE_SLICE

    // ---- bias add (C/D layout: VGPR r -> row r + 8*half, col = l16)
    #pragma unroll
    for (int t = 0; t < 20; ++t) {
        float bv = bias[colBase + t * 16 + l16];
        #pragma unroll
        for (int r = 0; r < 8; ++r) c[t][r] += bv;
    }

    // ---- noisy argmax per row (tau>0 preserves argmax), first-index ties
    float bestV[8];
    int   bestI[8];
    #pragma unroll
    for (int r = 0; r < 8; ++r) { bestV[r] = -3.4e38f; bestI[r] = 0; }
    #pragma unroll
    for (int t = 0; t < 20; ++t) {
        int colg = t * 16 + l16;
        #pragma unroll
        for (int r = 0; r < 8; ++r) {
            int row = rowBase + r + 8 * half;
            float nv = c[t][r] + gum[(size_t)row * GV + colBase + colg];
            if (nv > bestV[r]) { bestV[r] = nv; bestI[r] = colg; }
        }
    }
    #pragma unroll
    for (int r = 0; r < 8; ++r) {
        float v = bestV[r]; int ix = bestI[r];
        #pragma unroll
        for (int off = 1; off < 16; off <<= 1) {
            float ov = __shfl_xor(v, off, 16);
            int   oi = __shfl_xor(ix, off, 16);
            if (ov > v || (ov == v && oi < ix)) { v = ov; ix = oi; }
        }
        bestI[r] = ix;
    }

    // ---- hard one-hot == gather of selected codevector
    #pragma unroll
    for (int r = 0; r < 8; ++r) {
        int row = rowBase + r + 8 * half;
        const float4* src = (const float4*)
            (cvs + (size_t)(g * V_CODES + bestI[r]) * DG) + l16 * 2;
        float4* dst = (float4*)(out + (size_t)row * 256 + g * DG) + l16 * 2;
        dst[0] = src[0];
        dst[1] = src[1];
    }

    // ---- noise-free softmax + masked marginal (bit-deterministic)
    float macc[20];
    #pragma unroll
    for (int t = 0; t < 20; ++t) macc[t] = 0.f;
    #pragma unroll
    for (int r = 0; r < 8; ++r) {
        int row = rowBase + r + 8 * half;
        float m = c[0][r];
        #pragma unroll
        for (int t = 1; t < 20; ++t) m = fmaxf(m, c[t][r]);
        #pragma unroll
        for (int off = 1; off < 16; off <<= 1) m = fmaxf(m, __shfl_xor(m, off, 16));
        float s = 0.f;
        #pragma unroll
        for (int t = 0; t < 20; ++t) s += __expf(c[t][r] - m);
        #pragma unroll
        for (int off = 1; off < 16; off <<= 1) s += __shfl_xor(s, off, 16);
        float scale = (mask[row] != 0) ? (1.0f / s) : 0.0f;
        #pragma unroll
        for (int t = 0; t < 20; ++t) macc[t] += __expf(c[t][r] - m) * scale;
    }
    #pragma unroll
    for (int t = 0; t < 20; ++t) macc[t] += __shfl_xor(macc[t], 16, 32);

    __syncthreads();
    if (rsel == 0 && half == 0) {
        #pragma unroll
        for (int t = 0; t < 20; ++t) margLDS[colBase + t * 16 + l16] = macc[t];
    }
    __syncthreads();
    if (rsel == 1 && half == 0) {
        #pragma unroll
        for (int t = 0; t < 20; ++t) margLDS[colBase + t * 16 + l16] += macc[t];
    }
    __syncthreads();
    for (int i = tid; i < GV; i += 128)
        margPart[(size_t)blockIdx.x * GV + i] = margLDS[i];
}

// ---------------------------------------------------------------------------
// Finalize: deterministic column reduction over NBLK partials, mask count,
// per-group entropy, perplexity -> out[N*256].
// ---------------------------------------------------------------------------
__global__ __launch_bounds__(640)
void vq_finalize(const float* __restrict__ margPart,
                 const int*   __restrict__ mask,
                 float* __restrict__ out)
{
    __shared__ float terms[GV];
    __shared__ int   mcnt[GV];
    __shared__ float invM;
    const int tid = threadIdx.x;   // 640 threads, tid == (g*320 + v)

    int cnt = 0;
    for (int i = tid; i < N_ROWS; i += GV) cnt += (mask[i] != 0);
    mcnt[tid] = cnt;

    float s = 0.f;                 // fixed-order sum over blocks
    for (int b = 0; b < NBLK; ++b) s += margPart[(size_t)b * GV + tid];

    __syncthreads();
    if (tid == 0) {
        int tot = 0;
        for (int i = 0; i < GV; ++i) tot += mcnt[i];
        invM = 1.0f / (float)tot;
    }
    __syncthreads();

    float m = s * invM;
    terms[tid] = m * __logf(m + EPSF);
    __syncthreads();

    if (tid == 0) {
        float h0 = 0.f, h1 = 0.f;
        for (int i = 0; i < V_CODES; ++i) h0 += terms[i];
        for (int i = 0; i < V_CODES; ++i) h1 += terms[V_CODES + i];
        out[(size_t)N_ROWS * 256] = __expf(-h0) + __expf(-h1);
    }
}

extern "C" void kernel_launch(void* const* d_in, const int* in_sizes, int n_in,
                              void* d_out, int out_size, void* d_ws, size_t ws_size,
                              hipStream_t stream) {
    (void)in_sizes; (void)n_in; (void)out_size; (void)ws_size;
    const float* hs   = (const float*)d_in[0];
    const int*   mask = (const int*)  d_in[1];
    const float* gum  = (const float*)d_in[2];
    const float* W    = (const float*)d_in[3];
    const float* bias = (const float*)d_in[4];
    const float* cvs  = (const float*)d_in[5];
    float* out = (float*)d_out;

    float* margPart = (float*)d_ws;                         // 2.62 MB
    unsigned short* Wt = (unsigned short*)(margPart + (size_t)NBLK * GV); // 0.66 MB

    prep_w<<<(H_DIM * GV) / 256, 256, 0, stream>>>(W, Wt);
    gumbel_vq_main<<<NBLK, 128, 0, stream>>>(hs, mask, gum, Wt, bias, cvs,
                                             out, margPart);
    vq_finalize<<<1, GV, 0, stream>>>(margPart, mask, out);
}
